// PGAttention_18957985644958
// MI455X (gfx1250) — compile-verified, matrix operand order checked by reading, not run on hardware
//
#include <hip/hip_runtime.h>
#include <hip/hip_bf16.h>

// ---------------------------------------------------------------------------
// PGAttention for gfx1250 (MI455X): quantized attention with MSB gating mask.
// All matmuls run on V_WMMA_F32_16X16X32_F16 (exact for 2^-7 fixed-point data).
// GEMMs use 32x64 wave-tiles (8 WMMAs / 6 tile-loads per k-chunk); operand
// loads are grouped ahead of the WMMAs and the next chunk is prefetched with
// global_prefetch_b8 so latency is hidden without VGPR spills.
// ---------------------------------------------------------------------------

typedef __attribute__((ext_vector_type(16))) _Float16 v16h;
typedef __attribute__((ext_vector_type(8)))  _Float16 v8h;
typedef __attribute__((ext_vector_type(8)))  float    v8f;

#define BB 8
#define NN 1024
#define CC 768
#define HH 12
#define DD 64
#define STEP8    0.0078125f   // 2^(1-8)
#define INV8     128.0f
#define STEP4    0.125f       // 2^(1-4)
#define INV4     8.0f
#define MASK_TH  0.99f

__device__ __forceinline__ float quant_rne(float x, float inv_step, float step) {
    return rintf(x * inv_step) * step;   // v_rndne_f32 : round-to-nearest-even
}

// Load a 16x32 f16 tile (row-major, leading dim `ld`) into WMMA A-operand
// register layout (ISA 7.12.2): lane L<16 holds row L, K = kb..kb+7 and
// kb+16..kb+23 with kb = 8*(lane>=16). Two 16-byte loads per lane.
__device__ __forceinline__ v16h load_a16(const _Float16* tile, int ld, int lane) {
    const int row = lane & 15;
    const int kb  = (lane >> 4) << 3;
    const _Float16* p = tile + row * ld + kb;
    v8h lo = *(const v8h*)(p);
    v8h hi = *(const v8h*)(p + 16);
    return __builtin_shufflevector(lo, hi, 0,1,2,3,4,5,6,7,8,9,10,11,12,13,14,15);
}

__device__ __forceinline__ v8f wmma16(v16h a, v16h b, v8f c) {
    // 8 args: (neg_a, A, neg_b, B, c_mod, C, reuse_a, reuse_b)
    return __builtin_amdgcn_wmma_f32_16x16x32_f16(false, a, false, b, (short)0, c,
                                                  false, false);
}

// ---------------------------------------------------------------------------
// Elementwise fixed-point quantize f32 -> f16 (exact for this step size).
// ---------------------------------------------------------------------------
__global__ void PG_quant_f32_to_f16(const float* __restrict__ src,
                                    _Float16* __restrict__ dst,
                                    int n, float inv_step, float step) {
    int i = blockIdx.x * blockDim.x + threadIdx.x;
    const int stride = gridDim.x * blockDim.x;
    for (; i < n; i += stride)
        dst[i] = (_Float16)(rintf(src[i] * inv_step) * step);
}

// ---------------------------------------------------------------------------
// QKV GEMM: Y[8192,2304] = Xq @ Wqᵀ, fused epilogue scatters into per-head
// q / q_msb / k / k_msb (row-major [B,H,N,d]) and V transposed [B,H,d,N].
// One wave per 32x64 output tile (8 accumulators).
// ---------------------------------------------------------------------------
__global__ void PG_qkv_gemm(const _Float16* __restrict__ xq,
                            const _Float16* __restrict__ wq,
                            _Float16* __restrict__ q,  _Float16* __restrict__ qm,
                            _Float16* __restrict__ kq, _Float16* __restrict__ km,
                            _Float16* __restrict__ vt) {
    const int lane = threadIdx.x & 31;
    const int gw   = blockIdx.x * (blockDim.x >> 5) + (threadIdx.x >> 5);
    const int rowTile = gw & 255;        // 8192/32 tiles of M
    const int colT    = gw >> 8;         // 2304/64 tiles of N
    if (colT >= 36) return;
    const int n0  = rowTile << 5;
    const int cc0 = colT << 6;

    const _Float16* arow0 = xq + n0 * CC;
    const _Float16* arow1 = xq + (n0 + 16) * CC;
    const _Float16* brow0 = wq + (cc0 +  0) * CC;
    const _Float16* brow1 = wq + (cc0 + 16) * CC;
    const _Float16* brow2 = wq + (cc0 + 32) * CC;
    const _Float16* brow3 = wq + (cc0 + 48) * CC;

    v8f acc[2][4] = {};
    for (int kk = 0; kk < CC; kk += 32) {
        // warm the next chunk in L2/WGP$ (no VGPRs, no LOADcnt)
        __builtin_prefetch(arow0 + kk + 32, 0, 3);
        __builtin_prefetch(arow1 + kk + 32, 0, 3);
        __builtin_prefetch(brow0 + kk + 32, 0, 3);
        __builtin_prefetch(brow2 + kk + 32, 0, 3);
        // all 12 b128 loads for this chunk issue before the 8 WMMAs
        const v16h a0 = load_a16(arow0 + kk, CC, lane);
        const v16h a1 = load_a16(arow1 + kk, CC, lane);
        const v16h b0 = load_a16(brow0 + kk, CC, lane);
        const v16h b1 = load_a16(brow1 + kk, CC, lane);
        const v16h b2 = load_a16(brow2 + kk, CC, lane);
        const v16h b3 = load_a16(brow3 + kk, CC, lane);

        acc[0][0] = wmma16(a0, b0, acc[0][0]);
        acc[0][1] = wmma16(a0, b1, acc[0][1]);
        acc[0][2] = wmma16(a0, b2, acc[0][2]);
        acc[0][3] = wmma16(a0, b3, acc[0][3]);
        acc[1][0] = wmma16(a1, b0, acc[1][0]);
        acc[1][1] = wmma16(a1, b1, acc[1][1]);
        acc[1][2] = wmma16(a1, b2, acc[1][2]);
        acc[1][3] = wmma16(a1, b3, acc[1][3]);
    }

    const int row0 = (lane >> 4) << 3;   // C/D layout: lanes>=16 hold M=8..15
    const int cl   = lane & 15;
#pragma unroll
    for (int g = 0; g < 2; ++g) {
#pragma unroll
        for (int i = 0; i < 4; ++i) {
            const int cc = cc0 + i * 16 + cl;
            const int t  = cc / CC;          // 0:q 1:k 2:v
            const int r  = cc % CC;
            const int h  = r >> 6, dd = r & 63;
#pragma unroll
            for (int v = 0; v < 8; ++v) {
                const int n  = n0 + g * 16 + row0 + v;
                const int b  = n >> 10, nn = n & 1023;
                const int bh = b * HH + h;
                const float val = quant_rne(acc[g][i][v], INV8, STEP8);
                if (t == 0) {
                    const int idx = (bh * NN + nn) * DD + dd;
                    q[idx]  = (_Float16)val;
                    qm[idx] = (_Float16)quant_rne(val, INV4, STEP4);
                } else if (t == 1) {
                    const int idx = (bh * NN + nn) * DD + dd;
                    kq[idx] = (_Float16)val;
                    km[idx] = (_Float16)quant_rne(val, INV4, STEP4);
                } else {
                    vt[(bh * DD + dd) * NN + nn] = (_Float16)val;
                }
            }
        }
    }
}

// ---------------------------------------------------------------------------
// Attention: one block (4 waves) per (b,h, 16-query tile).
// LDS (160 KB of the 320 KB WGP pool):
//   sfull : 16x1024 f32 full-precision scaled logits (reused for PV partials)
//   smsb  : 16x1024 f32 MSB scaled logits
//   pbuf  : 16x1024 f16 quantized probabilities (WMMA A-operand source)
// ---------------------------------------------------------------------------
__global__ void PG_attention(const _Float16* __restrict__ q,
                             const _Float16* __restrict__ qm,
                             const _Float16* __restrict__ kq,
                             const _Float16* __restrict__ km,
                             const _Float16* __restrict__ vt,
                             _Float16* __restrict__ ao) {
    extern __shared__ char smem_raw[];
    float*    sfull = (float*)smem_raw;               // 64 KB
    float*    smsb  = sfull + 16 * NN;                // 64 KB
    _Float16* pbuf  = (_Float16*)(smsb + 16 * NN);    // 32 KB

    const int lane = threadIdx.x & 31;
    const int w    = threadIdx.x >> 5;                // wave 0..3
    const int qt   = blockIdx.x & 63;                 // query tile
    const int bh   = blockIdx.x >> 6;                 // 0..95
    const float scale = 0.125f;                       // d^-0.5 = 64^-0.5

    // --- Phase 1: S = Q Kᵀ (both precisions), each wave owns 256 keys -------
    const _Float16* qbase  = q  + (bh * NN + qt * 16) * DD;
    const _Float16* qmbase = qm + (bh * NN + qt * 16) * DD;
    const v16h aq0 = load_a16(qbase,      DD, lane);
    const v16h aq1 = load_a16(qbase + 32, DD, lane);
    const v16h am0 = load_a16(qmbase,      DD, lane);
    const v16h am1 = load_a16(qmbase + 32, DD, lane);

    const int row0 = (lane >> 4) << 3;
    const int cl   = lane & 15;
    for (int jt = w * 16; jt < w * 16 + 16; ++jt) {
        const int j0 = jt << 4;
        const _Float16* kbse = kq + (bh * NN + j0) * DD;
        const _Float16* kmse = km + (bh * NN + j0) * DD;
        // warm L2->WGP path for the next key tile (global_prefetch_b8)
        __builtin_prefetch(kbse + 16 * DD, 0, 3);
        __builtin_prefetch(kmse + 16 * DD, 0, 3);
        v8f zero = {};
        // B-operand = A-layout load of K rows x d  ==  Kᵀ tile
        v8f s  = wmma16(aq1, load_a16(kbse + 32, DD, lane),
                 wmma16(aq0, load_a16(kbse,      DD, lane), zero));
        v8f sm = wmma16(am1, load_a16(kmse + 32, DD, lane),
                 wmma16(am0, load_a16(kmse,      DD, lane), zero));
#pragma unroll
        for (int v = 0; v < 8; ++v) {
            sfull[(row0 + v) * NN + j0 + cl] = s[v]  * scale;
            smsb [(row0 + v) * NN + j0 + cl] = sm[v] * scale;
        }
    }
    __syncthreads();

    // --- Phase 2: per-row reductions. 8 lanes per row, 128 cols each --------
    const int row = (w << 2) + (lane & 3);            // wave owns 4 rows
    const int c0  = (lane >> 2) << 7;
    float m = -3.4e38f;
    for (int c = c0; c < c0 + 128; ++c) m = fmaxf(m, smsb[row * NN + c]);
    m = fmaxf(m, __shfl_xor(m, 4));
    m = fmaxf(m, __shfl_xor(m, 8));
    m = fmaxf(m, __shfl_xor(m, 16));
    float s = 0.f;
    for (int c = c0; c < c0 + 128; ++c) s += __expf(smsb[row * NN + c] - m);
    s += __shfl_xor(s, 4); s += __shfl_xor(s, 8); s += __shfl_xor(s, 16);
    // softmax(smsb) > TH  <=>  smsb > m + log(TH * sum)
    const float thr = m + __logf(MASK_TH * s);

    // masked full-precision logits (mask==0 -> logit 0, matching reference)
    float mf = -3.4e38f;
    for (int c = c0; c < c0 + 128; ++c) {
        const int idx = row * NN + c;
        const float fm = (smsb[idx] > thr) ? sfull[idx] : 0.0f;
        sfull[idx] = fm;
        mf = fmaxf(mf, fm);
    }
    mf = fmaxf(mf, __shfl_xor(mf, 4));
    mf = fmaxf(mf, __shfl_xor(mf, 8));
    mf = fmaxf(mf, __shfl_xor(mf, 16));
    float sf = 0.f;
    for (int c = c0; c < c0 + 128; ++c) sf += __expf(sfull[row * NN + c] - mf);
    sf += __shfl_xor(sf, 4); sf += __shfl_xor(sf, 8); sf += __shfl_xor(sf, 16);
    const float inv = 1.0f / sf;
    for (int c = c0; c < c0 + 128; ++c) {
        const float p = __expf(sfull[row * NN + c] - mf) * inv;
        pbuf[row * NN + c] = (_Float16)(rintf(p * INV8) * STEP8);  // quant(attn,8)
    }
    __syncthreads();

    // --- Phase 3: O = P V, each wave owns 256 keys, partials reduced in LDS -
    v8f o[4] = {};
    for (int j32 = 0; j32 < 8; ++j32) {
        const int j0 = (w * 8 + j32) << 5;
        __builtin_prefetch(vt + bh * DD * NN + j0 + 32, 0, 3);
        const v16h a = load_a16(pbuf + j0, NN, lane);          // LDS A-operand
#pragma unroll
        for (int dt = 0; dt < 4; ++dt) {
            // B-operand = A-layout load of Vᵀ rows(d) x keys  ==  V tile
            const v16h b = load_a16(vt + (bh * DD + dt * 16) * NN + j0, NN, lane);
            o[dt] = wmma16(a, b, o[dt]);
        }
    }
    float* part = sfull;                               // reuse: [4 waves][16][64]
#pragma unroll
    for (int dt = 0; dt < 4; ++dt)
#pragma unroll
        for (int v = 0; v < 8; ++v)
            part[w * 1024 + (row0 + v) * 64 + dt * 16 + cl] = o[dt][v];
    __syncthreads();

    // --- Phase 4: cross-wave sum, quant(out,8), store [B,N,C] --------------
    const int hh = bh % HH, b = bh / HH;
    const int dcol = w * 16 + cl;                      // wave w owns d cols w*16..
#pragma unroll
    for (int v = 0; v < 8; ++v) {
        const int r = row0 + v;
        const float sum = part[0 * 1024 + r * 64 + dcol]
                        + part[1 * 1024 + r * 64 + dcol]
                        + part[2 * 1024 + r * 64 + dcol]
                        + part[3 * 1024 + r * 64 + dcol];
        const int n = qt * 16 + r;
        ao[(b * NN + n) * CC + hh * DD + dcol] =
            (_Float16)quant_rne(sum, INV8, STEP8);
    }
}

// ---------------------------------------------------------------------------
// Projection GEMM: out[8192,768] = Aq @ Wpᵀ + bias  (f32 output)
// 32x64 wave-tile, same structure as QKV GEMM.
// ---------------------------------------------------------------------------
__global__ void PG_proj_gemm(const _Float16* __restrict__ aq,
                             const _Float16* __restrict__ wp,
                             const float* __restrict__ bias,
                             float* __restrict__ out) {
    const int lane = threadIdx.x & 31;
    const int gw   = blockIdx.x * (blockDim.x >> 5) + (threadIdx.x >> 5);
    const int rowTile = gw & 255;        // 8192/32
    const int colT    = gw >> 8;         // 768/64 = 12
    if (colT >= 12) return;
    const int n0  = rowTile << 5;
    const int cc0 = colT << 6;

    const _Float16* arow0 = aq + n0 * CC;
    const _Float16* arow1 = aq + (n0 + 16) * CC;
    const _Float16* brow0 = wp + (cc0 +  0) * CC;
    const _Float16* brow1 = wp + (cc0 + 16) * CC;
    const _Float16* brow2 = wp + (cc0 + 32) * CC;
    const _Float16* brow3 = wp + (cc0 + 48) * CC;

    v8f acc[2][4] = {};
    for (int kk = 0; kk < CC; kk += 32) {
        __builtin_prefetch(arow0 + kk + 32, 0, 3);
        __builtin_prefetch(arow1 + kk + 32, 0, 3);
        __builtin_prefetch(brow0 + kk + 32, 0, 3);
        __builtin_prefetch(brow2 + kk + 32, 0, 3);
        const v16h a0 = load_a16(arow0 + kk, CC, lane);
        const v16h a1 = load_a16(arow1 + kk, CC, lane);
        const v16h b0 = load_a16(brow0 + kk, CC, lane);
        const v16h b1 = load_a16(brow1 + kk, CC, lane);
        const v16h b2 = load_a16(brow2 + kk, CC, lane);
        const v16h b3 = load_a16(brow3 + kk, CC, lane);

        acc[0][0] = wmma16(a0, b0, acc[0][0]);
        acc[0][1] = wmma16(a0, b1, acc[0][1]);
        acc[0][2] = wmma16(a0, b2, acc[0][2]);
        acc[0][3] = wmma16(a0, b3, acc[0][3]);
        acc[1][0] = wmma16(a1, b0, acc[1][0]);
        acc[1][1] = wmma16(a1, b1, acc[1][1]);
        acc[1][2] = wmma16(a1, b2, acc[1][2]);
        acc[1][3] = wmma16(a1, b3, acc[1][3]);
    }

    const int row0 = (lane >> 4) << 3;
    const int cl   = lane & 15;
#pragma unroll
    for (int g = 0; g < 2; ++g) {
#pragma unroll
        for (int i = 0; i < 4; ++i) {
            const int cc = cc0 + i * 16 + cl;
            const float bv = bias[cc];
#pragma unroll
            for (int v = 0; v < 8; ++v)
                out[(n0 + g * 16 + row0 + v) * CC + cc] = acc[g][i][v] + bv;
        }
    }
}

// ---------------------------------------------------------------------------
extern "C" void kernel_launch(void* const* d_in, const int* in_sizes, int n_in,
                              void* d_out, int out_size, void* d_ws, size_t ws_size,
                              hipStream_t stream) {
    const float* x      = (const float*)d_in[0];   // [B,N,C]
    const float* w_qkv  = (const float*)d_in[1];   // [3C,C]
    const float* w_proj = (const float*)d_in[2];   // [C,C]
    const float* b_proj = (const float*)d_in[3];   // [C]
    float* out = (float*)d_out;                    // [B,N,C] f32

    // Workspace carve-up (f16 elements): ~93 MB total, fits L2-resident set.
    _Float16* ws = (_Float16*)d_ws;
    size_t off = 0;
    _Float16* xq = ws + off; off += (size_t)BB * NN * CC;        // quant8(x)
    _Float16* wq = ws + off; off += (size_t)3 * CC * CC;         // quant8(w_qkv)
    _Float16* wp = ws + off; off += (size_t)CC * CC;             // quant8(w_proj)
    _Float16* qb = ws + off; off += (size_t)BB * HH * NN * DD;   // q  [B,H,N,d]
    _Float16* qm = ws + off; off += (size_t)BB * HH * NN * DD;   // q_msb
    _Float16* kb = ws + off; off += (size_t)BB * HH * NN * DD;   // k
    _Float16* km = ws + off; off += (size_t)BB * HH * NN * DD;   // k_msb
    _Float16* vt = ws + off; off += (size_t)BB * HH * DD * NN;   // vᵀ [B,H,d,N]
    _Float16* ao = ws + off; off += (size_t)BB * NN * CC;        // attn out q8

    PG_quant_f32_to_f16<<<1024, 256, 0, stream>>>(x,      xq, BB * NN * CC, INV8, STEP8);
    PG_quant_f32_to_f16<<<1024, 256, 0, stream>>>(w_qkv,  wq, 3 * CC * CC,  INV8, STEP8);
    PG_quant_f32_to_f16<<<512,  256, 0, stream>>>(w_proj, wp, CC * CC,      INV8, STEP8);

    // 256 row-tiles(32) * 36 col-tiles(64) = 9216 waves / 4 per block
    PG_qkv_gemm<<<2304, 128, 0, stream>>>(xq, wq, qb, qm, kb, km, vt);

    // 96 (b,h) * 64 query tiles; 160 KB dynamic LDS per block (2 blocks/WGP)
    PG_attention<<<6144, 128, 160 * 1024, stream>>>(qb, qm, kb, km, vt, ao);

    // 256 * 12 = 3072 waves / 4 per block
    PG_proj_gemm<<<768, 128, 0, stream>>>(ao, wp, b_proj, out);
}
